// Attention_74199855005867
// MI455X (gfx1250) — matrix operand order repaired and checked by reading, not compiled, not run on hardware
//
#include <hip/hip_runtime.h>
#include <hip/hip_bf16.h>
#include <stdint.h>

// ---------------------------------------------------------------------------
// Problem constants (from reference)
// ---------------------------------------------------------------------------
#define BATCH   4
#define SEQ     2048
#define DIM     2048          // also N_HEADS*HEAD_DIM
#define NHEADS  16
#define HDIM    128
#define MROWS   (BATCH * SEQ) // 8192

typedef __attribute__((ext_vector_type(16))) __bf16 v16bf;
typedef __attribute__((ext_vector_type(8)))  __bf16 v8bf;
typedef __attribute__((ext_vector_type(8)))  float  v8f;
typedef __attribute__((ext_vector_type(4)))  unsigned int u32x4;
typedef __attribute__((ext_vector_type(8)))  int          i32x8;
typedef __attribute__((ext_vector_type(4)))  int          i32x4;

#if __has_builtin(__builtin_amdgcn_tensor_load_to_lds)
#define HAVE_TDM 1
#else
#define HAVE_TDM 0
#endif

__device__ __forceinline__ v8f wmma_bf16(v16bf a, v16bf b, v8f c) {
    // D = A(16x32 bf16) * B(32x16 bf16) + C(16x16 f32)
    return __builtin_amdgcn_wmma_f32_16x16x32_bf16(
        /*neg_a=*/false, a, /*neg_b=*/false, b,
        /*c_mod=*/(short)0, c, /*reuse_a=*/false, /*reuse_b=*/false);
}

// Async copy of one 16-byte chunk per lane: global -> LDS (ASYNCcnt-tracked).
__device__ __forceinline__ void async_copy_b128(uint32_t lds_off, const void* gptr) {
    asm volatile("global_load_async_to_lds_b128 %0, %1, off"
                 :: "v"(lds_off), "v"(gptr)
                 : "memory");
}
__device__ __forceinline__ void wait_async() {
    asm volatile("s_wait_asynccnt 0x0" ::: "memory");
}
__device__ __forceinline__ uint32_t lds_addr_of(const void* p) {
    return (uint32_t)(uintptr_t)p;   // LDS aperture: addr[31:0] = LDS offset
}

// ---------------------------------------------------------------------------
// Tensor Data Mover: 2D bf16 tile load, global -> LDS (TENSORcnt-tracked).
// tile0 = x extent (contiguous, elems), tile1 = y rows, stride0 in elems.
// D# packing per CDNA5 ISA 8.3/8.4 (count=1, type=2, data_size=2B, no pad,
// no iterate, no multicast; tensor dims == tile dims so no OOB clipping).
// ---------------------------------------------------------------------------
__device__ __forceinline__ void tdm_load_2d_bf16(uint32_t lds_off, const void* gaddr,
                                                 uint32_t tile0, uint32_t tile1,
                                                 uint32_t stride0) {
#if HAVE_TDM
    uint64_t ga = (uint64_t)(uintptr_t)gaddr;
    u32x4 g0;
    g0[0] = 1u;                                            // count=1 (valid)
    g0[1] = lds_off;                                       // lds_addr [63:32]
    g0[2] = (uint32_t)ga;                                  // global_addr lo
    g0[3] = (uint32_t)((ga >> 32) & 0x01FFFFFFu)           // global_addr[56:32]
            | 0x80000000u;                                 // type=2 ("image")
    i32x8 g1;
    g1[0] = (int)(1u << 16);                               // data_size=1 -> 2B
    g1[1] = (int)((tile0 & 0xFFFFu) << 16);                // tensor_dim0 lo16
    g1[2] = (int)((tile0 >> 16) | ((tile1 & 0xFFFFu) << 16)); // dim0 hi / dim1 lo
    g1[3] = (int)((tile1 >> 16) | (tile0 << 16));          // dim1 hi / tile_dim0
    g1[4] = (int)(tile1 & 0xFFFFu);                        // tile_dim1, tile_dim2=0
    g1[5] = (int)stride0;                                  // dim0_stride lo32
    g1[6] = 0;                                             // dim0_stride hi16
    g1[7] = 0;
    i32x4 gz = {0, 0, 0, 0};                               // 2D: groups 2/3 unused
#if __clang_major__ >= 23
    i32x8 gz8 = {0, 0, 0, 0, 0, 0, 0, 0};
    __builtin_amdgcn_tensor_load_to_lds(g0, g1, gz, gz, gz8, 0);
#else
    __builtin_amdgcn_tensor_load_to_lds(g0, g1, gz, gz, 0);
#endif
#else
    (void)lds_off; (void)gaddr; (void)tile0; (void)tile1; (void)stride0;
#endif
}
__device__ __forceinline__ void wait_tensor() {
#if HAVE_TDM
    __builtin_amdgcn_s_wait_tensorcnt(0);
#endif
}

// Load a 16x32 bf16 A-fragment from row-major memory.
// Layout: lanes 0-15 row=lane, K [0..15];
//         lanes 16-31 row=lane-16, K [16..31] (contiguous, matches B fills).
__device__ __forceinline__ v16bf load_a_frag(const __bf16* base_row, int half) {
    const __bf16* p = base_row + half * 16;
    v8bf lo = *(const v8bf*)(p);
    v8bf hi = *(const v8bf*)(p + 8);
    v16bf a;
#pragma unroll
    for (int i = 0; i < 8; ++i) { a[i] = lo[i]; a[8 + i] = hi[i]; }
    return a;
}

// ---------------------------------------------------------------------------
// 1) fp32 -> bf16 convert
// ---------------------------------------------------------------------------
__global__ void k_f32_to_bf16(const float* __restrict__ in,
                              __bf16* __restrict__ out, int n) {
    int i = blockIdx.x * blockDim.x + threadIdx.x;
    if (i < n) out[i] = (__bf16)in[i];
}

// 2) fp32 (K x N row-major) -> bf16 (N x K, i.e. column-major of original)
__global__ void k_transpose_bf16(const float* __restrict__ in,
                                 __bf16* __restrict__ out, int K, int N) {
    int i = blockIdx.x * blockDim.x + threadIdx.x;
    if (i >= K * N) return;
    int kk = i / N, nn = i - kk * N;      // coalesced read over n
    out[(size_t)nn * K + kk] = (__bf16)in[i];
}

// ---------------------------------------------------------------------------
// 3) GEMM: C(MxN) = A_bf16(MxK, row-major) * Bt_bf16(NxK, col-major of B)
//    Block: 256 threads = 8 waves, each wave -> 16x64 tile, block -> 128x64.
//    Shared B slab (64 cols x 32 K) staged in LDS by the Tensor Data Mover
//    (wave 0 issues one 2D-tile DMA per K step), double-buffered.
// ---------------------------------------------------------------------------
template <bool OUT_BF16>
__global__ __launch_bounds__(256)
void k_gemm_bf16(const __bf16* __restrict__ A,
                 const __bf16* __restrict__ Bt,
                 void* __restrict__ Cout,
                 int M, int N, int K) {
    __shared__ __align__(32) __bf16 bbuf[2][64 * 32];   // 2 x 4 KB

    const int lane = threadIdx.x & 31;
    const int wave = threadIdx.x >> 5;
    const int half = lane >> 4;
    const int l15  = lane & 15;
    const int tileM = blockIdx.y * 128 + wave * 16;
    const int tileN = blockIdx.x * 64;

    const __bf16* bslab = Bt + (size_t)tileN * K;       // col-major slab base
    const uint32_t ldsb[2] = { lds_addr_of(&bbuf[0][0]), lds_addr_of(&bbuf[1][0]) };

#if HAVE_TDM
    if (wave == 0)                      // prologue: slab for kb=0 -> buffer 0
        tdm_load_2d_bf16(ldsb[0], bslab, 32, 64, (uint32_t)K);
#else
    {   // fallback: per-lane async b128 copies
        const int col = threadIdx.x >> 2, c8 = threadIdx.x & 3;
        async_copy_b128(ldsb[0] + (uint32_t)(col * 32 + c8 * 8) * 2,
                        bslab + (size_t)col * K + c8 * 8);
    }
#endif

    v8f acc[4] = {};
    const __bf16* arow = A + (size_t)(tileM + l15) * K;

    for (int kb = 0; kb < K; kb += 32) {
        const int cur = (kb >> 5) & 1;
#if HAVE_TDM
        if (wave == 0) wait_tensor();   // issuing wave's DMA has landed
        __syncthreads();                // slab published to all waves
        if (kb + 32 < K && wave == 0)
            tdm_load_2d_bf16(ldsb[cur ^ 1], bslab + kb + 32, 32, 64, (uint32_t)K);
#else
        wait_async();
        __syncthreads();
        if (kb + 32 < K) {
            const int col = threadIdx.x >> 2, c8 = threadIdx.x & 3;
            async_copy_b128(ldsb[cur ^ 1] + (uint32_t)(col * 32 + c8 * 8) * 2,
                            bslab + (size_t)col * K + kb + 32 + c8 * 8);
        }
#endif
        v16bf a = load_a_frag(arow + kb, half);
        const __bf16* bb = &bbuf[cur][0];
#pragma unroll
        for (int t = 0; t < 4; ++t) {
            // B-fragment: lane = column, 16 contiguous K values (32B LDS read)
            v16bf b = *(const v16bf*)&bb[(t * 16 + l15) * 32 + half * 16];
            acc[t] = wmma_bf16(a, b, acc[t]);
        }
        __syncthreads();         // reads of bbuf[cur] done before next rewrite
    }

    // C layout: VGPR r -> row = 8*half + r, col = l15 (+ t*16)
    const int rowb = tileM + half * 8;
#pragma unroll
    for (int t = 0; t < 4; ++t) {
#pragma unroll
        for (int r = 0; r < 8; ++r) {
            size_t off = (size_t)(rowb + r) * N + tileN + t * 16 + l15;
            if (OUT_BF16) ((__bf16*)Cout)[off] = (__bf16)acc[t][r];
            else          ((float*) Cout)[off] = acc[t][r];
        }
    }
}

// ---------------------------------------------------------------------------
// 4) RoPE on q and k in place. One thread per (b,s,h,d<64) pair.
// ---------------------------------------------------------------------------
__global__ void k_rope(__bf16* __restrict__ q, __bf16* __restrict__ k,
                       const float* __restrict__ cosT,
                       const float* __restrict__ sinT) {
    int i = blockIdx.x * blockDim.x + threadIdx.x; // over B*S*H*64
    int d = i & 63;
    int h = (i >> 6) & (NHEADS - 1);
    int s = (i >> 10) & (SEQ - 1);
    size_t base = (size_t)(i >> 10) * DIM + h * HDIM + d; // (b*S+s)*DIM + ...
    float c  = cosT[s * 64 + d];
    float sn = sinT[s * 64 + d];

    float q1 = (float)q[base], q2 = (float)q[base + 64];
    q[base]      = (__bf16)(q1 * c - q2 * sn);
    q[base + 64] = (__bf16)(q2 * c + q1 * sn);

    float k1 = (float)k[base], k2 = (float)k[base + 64];
    k[base]      = (__bf16)(k1 * c - k2 * sn);
    k[base + 64] = (__bf16)(k2 * c + k1 * sn);
}

// ---------------------------------------------------------------------------
// 5) Flash attention (causal). Block = 8 waves = 128 q-rows of one (b,h).
//    K tile async-staged to LDS (ASYNCcnt path); V tile scalar-transposed.
// ---------------------------------------------------------------------------
__global__ __launch_bounds__(256)
void k_attn(const __bf16* __restrict__ q, const __bf16* __restrict__ k,
            const __bf16* __restrict__ v, __bf16* __restrict__ ctx) {
    __shared__ __align__(32) __bf16 ldsK[32 * HDIM];     // [key][dim]   8 KB
    __shared__ __align__(32) __bf16 ldsVt[HDIM * 32];    // [dim][key]   8 KB
    __shared__ __align__(32) __bf16 ldsP[8][16 * 32];    // per-wave P   8 KB

    const int tid  = threadIdx.x;
    const int lane = tid & 31;
    const int wave = tid >> 5;
    const int half = lane >> 4;
    const int l15  = lane & 15;

    const int b  = blockIdx.z;
    const int h  = blockIdx.y;
    const int q0 = blockIdx.x * 128;
    const int qrow = q0 + wave * 16;                 // this wave's 16 q rows
    const size_t bh = (size_t)b * SEQ * DIM + (size_t)h * HDIM;

    // Q A-fragments (constant across the key loop): 4 slabs over head dim.
    v16bf qa[4];
    {
        const __bf16* qp = q + bh + (size_t)(qrow + l15) * DIM;
#pragma unroll
        for (int t = 0; t < 4; ++t) qa[t] = load_a_frag(qp + t * 32, half);
    }

    float mrow[8], lrow[8];
#pragma unroll
    for (int r = 0; r < 8; ++r) { mrow[r] = -1e30f; lrow[r] = 0.0f; }
    v8f acc[8] = {};

    const float scale = 0.08838834764831845f;        // 1/sqrt(128)
    const int jend = q0 + 128;                       // causal upper bound

    for (int jb = 0; jb < jend; jb += 32) {
        __syncthreads();   // previous iteration's LDS reads are done
        // stage K tile [32 keys][128 dims] via async global->LDS (2 chunks/thread)
        for (int idx = tid; idx < 32 * 16; idx += 256) {
            int key = idx >> 4, c8 = idx & 15;
            async_copy_b128(lds_addr_of(&ldsK[key * HDIM + c8 * 8]),
                            k + bh + (size_t)(jb + key) * DIM + c8 * 8);
        }
        // stage V transposed [128 dims][32 keys] (scalar ds stores)
        for (int idx = tid; idx < 32 * HDIM; idx += 256) {
            int key = idx >> 7, d = idx & 127;
            ldsVt[d * 32 + key] = v[bh + (size_t)(jb + key) * DIM + d];
        }
        wait_async();      // this wave's K-tile chunks landed
        __syncthreads();   // all waves' chunks + V stores landed

        // scores: two 16x16 tiles (keys jb..jb+15 and jb+16..jb+31)
        v8f s0 = {}, s1 = {};
#pragma unroll
        for (int t = 0; t < 4; ++t) {
            v16bf b0 = *(const v16bf*)&ldsK[l15        * HDIM + t * 32 + half * 16];
            v16bf b1 = *(const v16bf*)&ldsK[(16 + l15) * HDIM + t * 32 + half * 16];
            s0 = wmma_bf16(qa[t], b0, s0);
            s1 = wmma_bf16(qa[t], b1, s1);
        }

        // online softmax (row = 8*half + r lives in one 16-lane half)
        const int qr_base = qrow + half * 8;
        const int j0 = jb + l15, j1 = jb + 16 + l15;
#pragma unroll
        for (int r = 0; r < 8; ++r) {
            const int qr = qr_base + r;
            float x0 = (j0 <= qr) ? s0[r] * scale : -1e30f;
            float x1 = (j1 <= qr) ? s1[r] * scale : -1e30f;
            float mx = fmaxf(x0, x1);
            for (int o = 8; o >= 1; o >>= 1) mx = fmaxf(mx, __shfl_xor(mx, o, 32));
            float mnew = fmaxf(mrow[r], mx);
            float f  = __expf(mrow[r] - mnew);
            float e0 = __expf(x0 - mnew);
            float e1 = __expf(x1 - mnew);
            float sum = e0 + e1;
            for (int o = 8; o >= 1; o >>= 1) sum += __shfl_xor(sum, o, 32);
            lrow[r] = lrow[r] * f + sum;
            mrow[r] = mnew;
#pragma unroll
            for (int t = 0; t < 8; ++t) acc[t][r] *= f;
            // stage P row into this wave's LDS patch (C layout -> memory)
            ldsP[wave][(half * 8 + r) * 32 + l15]      = (__bf16)e0;
            ldsP[wave][(half * 8 + r) * 32 + 16 + l15] = (__bf16)e1;
        }

        // re-read P as a 16x32 A-fragment (same wave; dscnt waits inserted)
        v16bf pa = load_a_frag(&ldsP[wave][l15 * 32], half);

        // ctx += P(16x32) * V(32x128)
#pragma unroll
        for (int t = 0; t < 8; ++t) {
            v16bf bv = *(const v16bf*)&ldsVt[(t * 16 + l15) * 32 + half * 16];
            acc[t] = wmma_bf16(pa, bv, acc[t]);
        }
    }

    // normalize and store context (bf16, layout (b*S+s, h*128+d))
#pragma unroll
    for (int r = 0; r < 8; ++r) {
        float inv = 1.0f / lrow[r];
        size_t row_off = bh + (size_t)(qrow + half * 8 + r) * DIM;
#pragma unroll
        for (int t = 0; t < 8; ++t)
            ctx[row_off + t * 16 + l15] = (__bf16)(acc[t][r] * inv);
    }
}

// ---------------------------------------------------------------------------
// Host launcher
// ---------------------------------------------------------------------------
extern "C" void kernel_launch(void* const* d_in, const int* in_sizes, int n_in,
                              void* d_out, int out_size, void* d_ws, size_t ws_size,
                              hipStream_t stream) {
    const float* x    = (const float*)d_in[0];
    const float* cosT = (const float*)d_in[1];
    const float* sinT = (const float*)d_in[2];
    const float* wq   = (const float*)d_in[3];
    const float* wk   = (const float*)d_in[4];
    const float* wv   = (const float*)d_in[5];
    const float* wo   = (const float*)d_in[6];

    char* ws = (char*)d_ws;
    const size_t SZ_X  = (size_t)MROWS * DIM * 2;     // 32 MB bf16
    const size_t SZ_W  = (size_t)DIM * DIM * 2;       //  8 MB bf16
    __bf16* xb   = (__bf16*)(ws);
    __bf16* wqT  = (__bf16*)(ws + SZ_X);
    __bf16* wkT  = (__bf16*)(ws + SZ_X + SZ_W);
    __bf16* wvT  = (__bf16*)(ws + SZ_X + 2 * SZ_W);
    __bf16* woT  = (__bf16*)(ws + SZ_X + 3 * SZ_W);
    __bf16* qb   = (__bf16*)(ws + SZ_X + 4 * SZ_W);
    __bf16* kb   = (__bf16*)(ws + 2 * SZ_X + 4 * SZ_W);
    __bf16* vb   = (__bf16*)(ws + 3 * SZ_X + 4 * SZ_W);
    __bf16* ctxb = (__bf16*)(ws + 4 * SZ_X + 4 * SZ_W);

    // 1) converts
    {
        int n = MROWS * DIM;
        k_f32_to_bf16<<<n / 256, 256, 0, stream>>>(x, xb, n);
        int nw = DIM * DIM;
        k_transpose_bf16<<<nw / 256, 256, 0, stream>>>(wq, wqT, DIM, DIM);
        k_transpose_bf16<<<nw / 256, 256, 0, stream>>>(wk, wkT, DIM, DIM);
        k_transpose_bf16<<<nw / 256, 256, 0, stream>>>(wv, wvT, DIM, DIM);
        k_transpose_bf16<<<nw / 256, 256, 0, stream>>>(wo, woT, DIM, DIM);
    }

    // 2) QKV projections
    dim3 gGemm(DIM / 64, MROWS / 128);
    k_gemm_bf16<true><<<gGemm, 256, 0, stream>>>(xb, wqT, qb, MROWS, DIM, DIM);
    k_gemm_bf16<true><<<gGemm, 256, 0, stream>>>(xb, wkT, kb, MROWS, DIM, DIM);
    k_gemm_bf16<true><<<gGemm, 256, 0, stream>>>(xb, wvT, vb, MROWS, DIM, DIM);

    // 3) RoPE on q,k
    {
        int n = BATCH * SEQ * NHEADS * 64;
        k_rope<<<n / 256, 256, 0, stream>>>(qb, kb, cosT, sinT);
    }

    // 4) causal flash attention
    dim3 gAttn(SEQ / 128, NHEADS, BATCH);
    k_attn<<<gAttn, 256, 0, stream>>>(qb, kb, vb, ctxb);

    // 5) output projection (fp32 out)
    k_gemm_bf16<false><<<gGemm, 256, 0, stream>>>(ctxb, woT, (float*)d_out,
                                                  MROWS, DIM, DIM);
}
